// UNetBottleneck_26474178412637
// MI455X (gfx1250) — matrix operand order repaired and checked by reading, not compiled, not run on hardware
//
#include <hip/hip_runtime.h>

typedef __attribute__((ext_vector_type(16))) _Float16 v16h;
typedef __attribute__((ext_vector_type(8)))  _Float16 v8h;
typedef __attribute__((ext_vector_type(8)))  float    v8f;
typedef int v4i_b __attribute__((vector_size(16)));   // matches builtin param type

#if defined(__HIP_DEVICE_COMPILE__) && defined(__has_builtin)
#if __has_builtin(__builtin_amdgcn_global_load_async_to_lds_b128) && \
    __has_builtin(__builtin_amdgcn_s_wait_asynccnt)
#define HAVE_ASYNC_LDS 1
#endif
#endif

// Problem dimensions (fixed by the reference's setup_inputs()).
constexpr int BATCH = 4, CIN = 48, C2 = 64, H = 368, W = 640;
constexpr int HW = H * W;
constexpr int PH = H + 2, PW = W + 2;                 // padded (halo) frame

// 9 taps x 2 input-channel groups (0..31, 32..47+pad) => K-chunks of 32.
constexpr int NCHUNK = 18;
constexpr int C1_FRAGS = 3 * NCHUNK;   // 48 outC -> 3 M-tiles
constexpr int C2_FRAGS = 4 * NCHUNK;   // 64 outC -> 4 M-tiles
constexpr int M1_FRAGS = 8 * 2;        // mlp1: M=128 (8 tiles), K=64 (2 chunks)
constexpr int M2_FRAGS = 1 * 4;        // mlp2: M=10->16 (1 tile), K=128 (4 chunks)
constexpr int TOTAL_FRAGS = C1_FRAGS + C2_FRAGS + M1_FRAGS + M2_FRAGS; // 146
constexpr int STAGE2_FRAGS = C2_FRAGS + M1_FRAGS + M2_FRAGS;           // 92

// Workspace layout (_Float16 elements):
//   [0, NHWC_ELEMS)                input transposed to padded NHWC f16
//   [NHWC_ELEMS, 2*NHWC_ELEMS)     conv1 output "h", padded NHWC f16
//   [2*NHWC_ELEMS, ...)            packed WMMA A-fragments (512 halves each)
constexpr long long PPIX       = (long long)BATCH * PH * PW;      // 950,160
constexpr long long NHWC_ELEMS = PPIX * CIN;                      // 45,607,680
constexpr long long H_OFF      = NHWC_ELEMS;
constexpr long long FRAG_BASE  = 2 * NHWC_ELEMS;
constexpr long long C1_OFF = FRAG_BASE;
constexpr long long C2_OFF = C1_OFF + (long long)C1_FRAGS * 512;

constexpr long long CONV_OUT_ELEMS = (long long)BATCH * C2 * HW;  // 60,293,120

__device__ __forceinline__ float swapx16(float x) {
  // ds_swizzle group-of-32: xor_mask=0x10, and_mask=0x1f -> lane <-> lane^16
  return __int_as_float(__builtin_amdgcn_ds_swizzle(__float_as_int(x), 0x401F));
}

__device__ __forceinline__ v8f wmma_f16(v16h a, v16h b, v8f c) {
  return __builtin_amdgcn_wmma_f32_16x16x32_f16(false, a, false, b, (short)0, c,
                                                false, false);
}

// Cooperative: copy `halves` _Float16 from global ws into LDS, 16B per issue.
// Uses the gfx1250 async-to-LDS path (ASYNCcnt) when available.
__device__ __forceinline__ void stage_to_lds(const _Float16* gsrc,
                                             _Float16* lds, int halves) {
  const int tid = threadIdx.x;
#ifdef HAVE_ASYNC_LDS
  auto gp = (__attribute__((address_space(1))) v4i_b*)gsrc;   // 16B units, global
  auto lp = (__attribute__((address_space(3))) v4i_b*)lds;    // 16B units, LDS
  const int n16 = halves / 8;
  for (int i = tid; i < n16; i += 128)
    __builtin_amdgcn_global_load_async_to_lds_b128(gp + i, lp + i, 0, 0);
  __builtin_amdgcn_s_wait_asynccnt(0);
  __syncthreads();
#else
  for (int idx = tid * 8; idx < halves; idx += 128 * 8)
    *(v8h*)(lds + idx) = *(const v8h*)(gsrc + idx);
  __syncthreads();
#endif
}

// ---------------------------------------------------------------------------
// Pack all weights into WMMA A-fragment-native f16 layout.
// A fragment (16x32 f16): lane = (m, hi), half j -> k = hi*16 + j (contiguous order chosen here; B uses the same).
// ---------------------------------------------------------------------------
__global__ void pack_frags(const float* __restrict__ w1, const float* __restrict__ w2,
                           const float* __restrict__ mw1, const float* __restrict__ mw2,
                           _Float16* __restrict__ ws) {
  int tid = blockIdx.x * blockDim.x + threadIdx.x;
  if (tid >= TOTAL_FRAGS * 512) return;
  int f    = tid >> 9;
  int r    = tid & 511;
  int lane = r >> 4, j = r & 15;
  int m = lane & 15, hi = lane >> 4;
  int kl = hi * 16 + j;   // K index within the 32-chunk

  float val = 0.f;
  if (f < C1_FRAGS) {                                  // conv1: w1 (48,48,3,3)
    int mt = f / NCHUNK, ch = f % NCHUNK;
    int tap = ch >> 1, g = ch & 1;
    int ic = g * 32 + kl;
    if (ic < CIN) val = w1[((mt * 16 + m) * CIN + ic) * 9 + tap];
  } else if (f < C1_FRAGS + C2_FRAGS) {                // conv2: w2 (64,48,3,3)
    int fl = f - C1_FRAGS;
    int mt = fl / NCHUNK, ch = fl % NCHUNK;
    int tap = ch >> 1, g = ch & 1;
    int ic = g * 32 + kl;
    if (ic < CIN) val = w2[((mt * 16 + m) * CIN + ic) * 9 + tap];
  } else if (f < C1_FRAGS + C2_FRAGS + M1_FRAGS) {     // mlp1: mw1 (128,64)
    int fl = f - C1_FRAGS - C2_FRAGS;
    int mt = fl >> 1, kc = fl & 1;
    val = mw1[(mt * 16 + m) * 64 + kc * 32 + kl];
  } else {                                             // mlp2: mw2 (10,128), pad M to 16
    int kc = f - (C1_FRAGS + C2_FRAGS + M1_FRAGS);
    if (m < 10) val = mw2[m * 128 + kc * 32 + kl];
  }
  ws[FRAG_BASE + (long long)f * 512 + lane * 16 + j] = (_Float16)val;
}

// ---------------------------------------------------------------------------
// Transpose input NCHW f32 -> padded NHWC f16 (halo zero-filled).
// ---------------------------------------------------------------------------
__global__ void transpose_in(const float* __restrict__ in, _Float16* __restrict__ ws) {
  long long t = (long long)blockIdx.x * blockDim.x + threadIdx.x;
  if (t >= PPIX) return;
  int px = (int)(t % PW);
  long long r = t / PW;
  int py = (int)(r % PH);
  int b  = (int)(r / PH);

  v16h v0 = {}, v1 = {}, v2 = {};
  if (py >= 1 && py <= H && px >= 1 && px <= W) {
    const float* src = in + (long long)b * CIN * HW + (py - 1) * W + (px - 1);
#pragma unroll
    for (int c = 0; c < 16; ++c) v0[c] = (_Float16)src[(long long)c * HW];
#pragma unroll
    for (int c = 0; c < 16; ++c) v1[c] = (_Float16)src[(long long)(c + 16) * HW];
#pragma unroll
    for (int c = 0; c < 16; ++c) v2[c] = (_Float16)src[(long long)(c + 32) * HW];
  }
  _Float16* dst = ws + t * CIN;
  *(v16h*)(dst)      = v0;
  *(v16h*)(dst + 16) = v1;
  *(v16h*)(dst + 32) = v2;
}

// Zero the halo ring of the "h" buffer (interior is written by conv1).
__global__ void zero_h_border(_Float16* __restrict__ ws) {
  long long t = (long long)blockIdx.x * blockDim.x + threadIdx.x;
  if (t >= PPIX) return;
  int px = (int)(t % PW);
  long long r = t / PW;
  int py = (int)(r % PH);
  if (py >= 1 && py <= H && px >= 1 && px <= W) return;
  _Float16* dst = ws + H_OFF + t * CIN;
  v16h z = {};
  *(v16h*)(dst)      = z;
  *(v16h*)(dst + 16) = z;
  *(v16h*)(dst + 32) = z;
}

// ---------------------------------------------------------------------------
// Conv1: h = relu(conv3x3(in_nhwc, w1) + b1), written as padded NHWC f16.
// A-fragments staged once per block into LDS (async); B fragment is one
// aligned 32B load with the tap offset folded into the immediate.
// ---------------------------------------------------------------------------
__global__ void __launch_bounds__(128)
conv1_kernel(const float* __restrict__ b1, _Float16* __restrict__ ws) {
  __shared__ __align__(32) _Float16 ldsA[C1_FRAGS * 512];   // 55,296 B
  stage_to_lds(ws + C1_OFF, ldsA, C1_FRAGS * 512);

  const int lane = threadIdx.x & 31;
  const int wave = threadIdx.x >> 5;
  const int n = lane & 15, hi = lane >> 4;
  const int tilesX = W / 64;                       // 10
  int bid = blockIdx.x;
  int x0 = (bid % tilesX) * 64 + wave * 16;
  int y  = (bid / tilesX) % H;
  int b  = bid / (tilesX * H);
  const int x = x0 + n;

  const long long pixOff = ((long long)(b * PH + y + 1) * PW + (x + 1)) * CIN;
  const _Float16* pixB = ws + pixOff + hi * 16;
  const _Float16* frag = ldsA + lane * 16;

  v8f acc[3] = {};
#pragma unroll
  for (int ch = 0; ch < NCHUNK; ++ch) {
    const int tap = ch >> 1, g = ch & 1;
    const int dy = tap / 3 - 1, dx = tap % 3 - 1;
    v16h bf = *(const v16h*)(pixB + (dy * PW + dx) * CIN + g * 32);
#pragma unroll
    for (int mt = 0; mt < 3; ++mt) {
      v16h a = *(const v16h*)(frag + (mt * NCHUNK + ch) * 512);
      acc[mt] = wmma_f16(a, bf, acc[mt]);
    }
  }

  _Float16* hp = ws + H_OFF + pixOff;
#pragma unroll
  for (int mt = 0; mt < 3; ++mt) {
    v8h o;
#pragma unroll
    for (int v = 0; v < 8; ++v) {
      int c = mt * 16 + hi * 8 + v;                 // D layout: (m = hi*8+v, n)
      o[v] = (_Float16)fmaxf(acc[mt][v] + b1[c], 0.f);
    }
    *(v8h*)(hp + mt * 16 + hi * 8) = o;
  }
}

// ---------------------------------------------------------------------------
// Conv2 + per-pixel MLP + softmax/sigmoid + per-pixel filter + warp + blend.
// All weight fragments (conv2 + mlp1 + mlp2) staged in LDS once per block;
// conv D-tiles are recycled in-register into MLP B-fragments via SWAPX16.
// ---------------------------------------------------------------------------
__global__ void __launch_bounds__(128)
conv2_mlp_kernel(const _Float16* __restrict__ ws,
                 const float* __restrict__ b2,  const float* __restrict__ mb1,
                 const float* __restrict__ mb2, const float* __restrict__ rad,
                 const float* __restrict__ prev, const float* __restrict__ mv,
                 float* __restrict__ out) {
  __shared__ __align__(32) _Float16 ldsA[STAGE2_FRAGS * 512];  // 94,208 B
  stage_to_lds(ws + C2_OFF, ldsA, STAGE2_FRAGS * 512);

  const int lane = threadIdx.x & 31;
  const int wave = threadIdx.x >> 5;
  const int n = lane & 15, hi = lane >> 4;
  const int tilesX = W / 64;
  int bid = blockIdx.x;
  int x0 = (bid % tilesX) * 64 + wave * 16;
  int y  = (bid / tilesX) % H;
  int b  = bid / (tilesX * H);
  const int x = x0 + n;

  const long long pixOff = ((long long)(b * PH + y + 1) * PW + (x + 1)) * CIN;
  const _Float16* pixB   = ws + H_OFF + pixOff + hi * 16;
  const _Float16* fragC  = ldsA + lane * 16;
  const _Float16* fragM1 = ldsA + C2_FRAGS * 512 + lane * 16;
  const _Float16* fragM2 = ldsA + (C2_FRAGS + M1_FRAGS) * 512 + lane * 16;

  // ---- conv2: 18 chunks x 4 M-tiles ----
  v8f acc[4] = {};
#pragma unroll
  for (int ch = 0; ch < NCHUNK; ++ch) {
    const int tap = ch >> 1, g = ch & 1;
    const int dy = tap / 3 - 1, dx = tap % 3 - 1;
    v16h bf = *(const v16h*)(pixB + (dy * PW + dx) * CIN + g * 32);
#pragma unroll
    for (int mt = 0; mt < 4; ++mt) {
      v16h a = *(const v16h*)(fragC + (mt * NCHUNK + ch) * 512);
      acc[mt] = wmma_f16(a, bf, acc[mt]);
    }
  }

  // bias + relu, emit "conv" output (first tuple element, NCHW f32)
#pragma unroll
  for (int mt = 0; mt < 4; ++mt) {
#pragma unroll
    for (int v = 0; v < 8; ++v) {
      int c = mt * 16 + hi * 8 + v;
      float o = fmaxf(acc[mt][v] + b2[c], 0.f);
      acc[mt][v] = o;
      out[(long long)(b * C2 + c) * HW + y * W + x] = o;
    }
  }

  // ---- D-tiles -> B-fragments for GEMM1 (K=64 over conv channels) ----
  v16h xb[2];
#pragma unroll
  for (int kc = 0; kc < 2; ++kc) {
#pragma unroll
    for (int j = 0; j < 8; ++j) {
      float s0 = swapx16(acc[2 * kc][j]);       // partner's tile 2kc, m=8+j (for hi=0)
      float s1 = swapx16(acc[2 * kc + 1][j]);   // partner's tile 2kc+1, m=j   (for hi=1)
      float lo = hi ? s1 : acc[2 * kc][j];      // k_local = hi*16 + j
      float up = hi ? acc[2 * kc + 1][j] : s0;  // k_local = hi*16 + 8 + j
      xb[kc][j]     = (_Float16)lo;
      xb[kc][j + 8] = (_Float16)up;
    }
  }

  // ---- GEMM1: (128x64) x (64x16) ----
  v8f hacc[8] = {};
#pragma unroll
  for (int mt = 0; mt < 8; ++mt) {
#pragma unroll
    for (int kc = 0; kc < 2; ++kc) {
      v16h a = *(const v16h*)(fragM1 + (mt * 2 + kc) * 512);
      hacc[mt] = wmma_f16(a, xb[kc], hacc[mt]);
    }
  }
#pragma unroll
  for (int mt = 0; mt < 8; ++mt)
#pragma unroll
    for (int v = 0; v < 8; ++v) {
      int d = mt * 16 + hi * 8 + v;
      hacc[mt][v] = fmaxf(hacc[mt][v] + mb1[d], 0.f);
    }

  // ---- hidden D-tiles -> B-fragments for GEMM2 (K=128) ----
  v16h hb[4];
#pragma unroll
  for (int kc = 0; kc < 4; ++kc) {
#pragma unroll
    for (int j = 0; j < 8; ++j) {
      float s0 = swapx16(hacc[2 * kc][j]);
      float s1 = swapx16(hacc[2 * kc + 1][j]);
      float lo = hi ? s1 : hacc[2 * kc][j];
      float up = hi ? hacc[2 * kc + 1][j] : s0;
      hb[kc][j]     = (_Float16)lo;
      hb[kc][j + 8] = (_Float16)up;
    }
  }

  // ---- GEMM2: (16x128, rows>=10 zero) x (128x16) ----
  v8f oacc = {};
#pragma unroll
  for (int kc = 0; kc < 4; ++kc) {
    v16h a = *(const v16h*)(fragM2 + kc * 512);
    oacc = wmma_f16(a, hb[kc], oacc);
  }
#pragma unroll
  for (int v = 0; v < 8; ++v) {
    int d = hi * 8 + v;
    oacc[v] += (d < 10) ? mb2[d] : 0.f;
  }

  // lane (n,0) holds ch0..7; fetch ch8, ch9 from partner lane (n,1)
  float c8 = swapx16(oacc[0]);
  float c9 = swapx16(oacc[1]);

  if (hi == 0) {
    float kv[9];
#pragma unroll
    for (int t = 0; t < 8; ++t) kv[t] = oacc[t];
    kv[8] = c8;
    float alpha = c9;

    float mx = kv[0];
#pragma unroll
    for (int t = 1; t < 9; ++t) mx = fmaxf(mx, kv[t]);
    float e[9], s = 0.f;
#pragma unroll
    for (int t = 0; t < 9; ++t) { e[t] = __expf(kv[t] - mx); s += e[t]; }
    float inv = 1.f / s;
    float at  = 1.f / (1.f + __expf(-alpha));

    // per-pixel 3x3 filter of radiance (zero-padded)
    float o[3] = {0.f, 0.f, 0.f};
#pragma unroll
    for (int t = 0; t < 9; ++t) {
      int yy = y + t / 3 - 1, xx = x + t % 3 - 1;
      if (yy >= 0 && yy < H && xx >= 0 && xx < W) {
        float wgt = e[t] * inv;
#pragma unroll
        for (int c = 0; c < 3; ++c)
          o[c] += wgt * rad[(long long)(b * 3 + c) * HW + yy * W + xx];
      }
    }

    // nearest warp of prev_filtered: clip then truncate (matches astype(int32))
    float mvx = mv[(long long)(b * 2 + 0) * HW + y * W + x];
    float mvy = mv[(long long)(b * 2 + 1) * HW + y * W + x];
    int wi = (int)fminf(fmaxf((float)y + mvy, 0.f), (float)(H - 1));
    int wj = (int)fminf(fmaxf((float)x + mvx, 0.f), (float)(W - 1));

    float* fout = out + CONV_OUT_ELEMS;
#pragma unroll
    for (int c = 0; c < 3; ++c) {
      float pw = prev[(long long)(b * 3 + c) * HW + wi * W + wj];
      fout[(long long)(b * 3 + c) * HW + y * W + x] = o[c] * (1.f - at) + pw * at;
    }
  }
}

extern "C" void kernel_launch(void* const* d_in, const int* in_sizes, int n_in,
                              void* d_out, int out_size, void* d_ws, size_t ws_size,
                              hipStream_t stream) {
  const float* input    = (const float*)d_in[0];
  const float* radiance = (const float*)d_in[1];
  const float* prev     = (const float*)d_in[2];
  const float* motion   = (const float*)d_in[3];
  const float* w1       = (const float*)d_in[4];
  const float* b1       = (const float*)d_in[5];
  const float* w2       = (const float*)d_in[6];
  const float* b2       = (const float*)d_in[7];
  const float* mw1      = (const float*)d_in[8];
  const float* mb1      = (const float*)d_in[9];
  const float* mw2      = (const float*)d_in[10];
  const float* mb2      = (const float*)d_in[11];
  float*    out = (float*)d_out;
  _Float16* ws  = (_Float16*)d_ws;

  pack_frags<<<(TOTAL_FRAGS * 512 + 255) / 256, 256, 0, stream>>>(w1, w2, mw1, mw2, ws);

  const int pblk = (int)((PPIX + 127) / 128);
  transpose_in <<<pblk, 128, 0, stream>>>(input, ws);
  zero_h_border<<<pblk, 128, 0, stream>>>(ws);

  const int grid = (W / 64) * H * BATCH;   // 14720 blocks, 4 waves each
  conv1_kernel<<<grid, 128, 0, stream>>>(b1, ws);
  conv2_mlp_kernel<<<grid, 128, 0, stream>>>(ws, b2, mb1, mb2,
                                             radiance, prev, motion, out);
}
